// TransformerLayerShard_46488726012682
// MI455X (gfx1250) — matrix operand order repaired and checked by reading, not compiled, not run on hardware
//
#include <hip/hip_runtime.h>

// ---------------------------------------------------------------------------
// Types / helpers
// ---------------------------------------------------------------------------
typedef __bf16 bf16x16 __attribute__((ext_vector_type(16)));
typedef float  f32x8   __attribute__((ext_vector_type(8)));
typedef int    v4i     __attribute__((vector_size(16)));   // matches builtin proto
typedef unsigned short u16;

#define SEQ    2048
#define DMODEL 4096
#define NHEADS 16
#define DHEAD  256
#define ROT    64
#define DFF    16384

#if defined(__has_builtin)
#if __has_builtin(__builtin_amdgcn_global_load_async_to_lds_b128) && \
    __has_builtin(__builtin_amdgcn_s_wait_asynccnt)
#define HAVE_ASYNC_LDS 1
#endif
#endif

#ifndef HAVE_ASYNC_LDS
#define HAVE_ASYNC_LDS 0
#endif

#define GAS1(p) ((__attribute__((address_space(1))) v4i*)(p))
#define LAS3(p) ((__attribute__((address_space(3))) v4i*)(p))

union FragAB { bf16x16 v; uint4 q[2]; };

__device__ __forceinline__ u16 f2bf(float f) {
    unsigned int u = __float_as_uint(f);
    unsigned int r = u + 0x7fffu + ((u >> 16) & 1u);   // round-to-nearest-even
    return (u16)(r >> 16);
}
__device__ __forceinline__ float bf2f(u16 h) {
    return __uint_as_float(((unsigned int)h) << 16);
}
__device__ __forceinline__ float geluf(float x) {
    const float c = 0.7978845608028654f; // sqrt(2/pi)
    float t = tanhf(c * (x + 0.044715f * x * x * x));
    return 0.5f * x * (1.0f + t);
}

// ---------------------------------------------------------------------------
// LayerNorm: x[SEQ][DMODEL] f32 -> xn bf16
// ---------------------------------------------------------------------------
__global__ __launch_bounds__(256) void ln_kernel(const float* __restrict__ x,
                                                 const float* __restrict__ g,
                                                 const float* __restrict__ b,
                                                 u16* __restrict__ xn) {
    const int row = blockIdx.x;
    const float* xr = x + (size_t)row * DMODEL;
    float s = 0.f, s2 = 0.f;
    float v[16];
#pragma unroll
    for (int j = 0; j < 16; ++j) {
        v[j] = xr[threadIdx.x + j * 256];
        s  += v[j];
        s2 += v[j] * v[j];
    }
    __shared__ float rs[256], rq[256];
    rs[threadIdx.x] = s; rq[threadIdx.x] = s2;
    __syncthreads();
    for (int off = 128; off > 0; off >>= 1) {
        if (threadIdx.x < off) {
            rs[threadIdx.x] += rs[threadIdx.x + off];
            rq[threadIdx.x] += rq[threadIdx.x + off];
        }
        __syncthreads();
    }
    const float mean = rs[0] * (1.0f / DMODEL);
    const float var  = rq[0] * (1.0f / DMODEL) - mean * mean;
    const float rstd = rsqrtf(var + 1e-5f);
#pragma unroll
    for (int j = 0; j < 16; ++j) {
        int c = threadIdx.x + j * 256;
        float o = (v[j] - mean) * rstd * g[c] + b[c];
        xn[(size_t)row * DMODEL + c] = f2bf(o);
    }
}

// ---------------------------------------------------------------------------
// Weight convert + transpose: W[K][N] f32 -> WT[N][K] bf16
// ---------------------------------------------------------------------------
__global__ __launch_bounds__(256) void wtr_f32_bf16(const float* __restrict__ W,
                                                    u16* __restrict__ WT,
                                                    int K, int N) {
    __shared__ float t[32][33];
    const int n0 = blockIdx.x * 32, k0 = blockIdx.y * 32;
    const int c = threadIdx.x & 31, r8 = threadIdx.x >> 5;
#pragma unroll
    for (int i = 0; i < 4; ++i) {
        int r = r8 + i * 8;
        t[r][c] = W[(size_t)(k0 + r) * N + n0 + c];
    }
    __syncthreads();
#pragma unroll
    for (int i = 0; i < 4; ++i) {
        int r = r8 + i * 8;
        WT[(size_t)(n0 + r) * K + k0 + c] = f2bf(t[c][r]);
    }
}

// Generic bf16 tile transpose (batched):  D[b][c][r] = S[b][r][c]
__global__ __launch_bounds__(256) void tr_u16(const u16* __restrict__ S, long long lds_,
                                              long long sbs,
                                              u16* __restrict__ D, long long ldd,
                                              long long dbs) {
    S += (size_t)blockIdx.z * sbs;
    D += (size_t)blockIdx.z * dbs;
    __shared__ u16 t[32][33];
    const int c0 = blockIdx.x * 32, r0 = blockIdx.y * 32;
    const int c = threadIdx.x & 31, r8 = threadIdx.x >> 5;
#pragma unroll
    for (int i = 0; i < 4; ++i) {
        int r = r8 + i * 8;
        t[r][c] = S[(size_t)(r0 + r) * lds_ + c0 + c];
    }
    __syncthreads();
#pragma unroll
    for (int i = 0; i < 4; ++i) {
        int r = r8 + i * 8;
        D[(size_t)(c0 + r) * ldd + r0 + c] = t[c][r];
    }
}

// ---------------------------------------------------------------------------
// RoPE on first ROT dims of each head (bf16 in -> bf16 out)
// ---------------------------------------------------------------------------
__global__ __launch_bounds__(256) void rope_kernel(const u16* __restrict__ src,
                                                   u16* __restrict__ dst) {
    const size_t idx = (size_t)blockIdx.x * 256 + threadIdx.x;
    const int t = (int)(idx >> 12);          // /4096
    const int c = (int)(idx & 4095);
    const int d = c & (DHEAD - 1);
    float v = bf2f(src[idx]);
    float o;
    if (d < ROT) {
        const float inv = __expf(-((float)(d & ~1) / (float)ROT) * 9.210340371976184f); // ln(1e4)
        const float ang = (float)t * inv;
        const float sn = __sinf(ang), cs = __cosf(ang);
        const float partner = bf2f(src[idx ^ 1]);
        o = (d & 1) ? (v * cs + partner * sn) : (v * cs - partner * sn);
    } else {
        o = v;
    }
    dst[idx] = f2bf(o);
}

// ---------------------------------------------------------------------------
// Row softmax over logits [NHEADS*SEQ][SEQ] f32 -> probs bf16
// ---------------------------------------------------------------------------
__global__ __launch_bounds__(256) void softmax_kernel(const float* __restrict__ logits,
                                                      u16* __restrict__ probs) {
    const size_t row = blockIdx.x;
    const float* lr = logits + row * SEQ;
    float v[8];
    float mx = -3.0e38f;
#pragma unroll
    for (int j = 0; j < 8; ++j) {
        v[j] = lr[threadIdx.x + j * 256];
        mx = fmaxf(mx, v[j]);
    }
    __shared__ float red[256];
    red[threadIdx.x] = mx;
    __syncthreads();
    for (int off = 128; off > 0; off >>= 1) {
        if (threadIdx.x < off) red[threadIdx.x] = fmaxf(red[threadIdx.x], red[threadIdx.x + off]);
        __syncthreads();
    }
    mx = red[0];
    __syncthreads();
    float s = 0.f;
#pragma unroll
    for (int j = 0; j < 8; ++j) { v[j] = __expf(v[j] - mx); s += v[j]; }
    red[threadIdx.x] = s;
    __syncthreads();
    for (int off = 128; off > 0; off >>= 1) {
        if (threadIdx.x < off) red[threadIdx.x] += red[threadIdx.x + off];
        __syncthreads();
    }
    const float inv = 1.0f / red[0];
#pragma unroll
    for (int j = 0; j < 8; ++j)
        probs[row * SEQ + threadIdx.x + j * 256] = f2bf(v[j] * inv);
}

// ---------------------------------------------------------------------------
// Tiled bf16 WMMA GEMM:  C[m][n] = sum_k A[m][k] * B[n][k]   (B stored [N][K])
// Tile 128x128x32, 256 threads = 8 wave32, each wave 32x64 (8 accumulators).
// Double-buffered LDS; async global->LDS copies when the toolchain has them.
// Epilogues: 0 = f32 store, 1 = bf16 store,
//            2 = logits (scale + bias2d[m][n] + causal mask, f32),
//            3 = bias[n] + GELU -> bf16, 4 = C += acc + bias[n] (f32 RMW)
// ---------------------------------------------------------------------------
__global__ __launch_bounds__(256) void gemm_bf16(const u16* __restrict__ A, long long lda, long long sA,
                                                 const u16* __restrict__ B, long long ldb, long long sB,
                                                 void* __restrict__ C, long long ldc, long long sC,
                                                 int K, int epi,
                                                 const float* __restrict__ bias, float scale) {
    const int tid  = threadIdx.x;
    const int lane = tid & 31;
    const int wid  = tid >> 5;
    const int wm   = wid & 3;          // 0..3  -> 32-row group
    const int wn   = wid >> 2;         // 0..1  -> 64-col group
    const int hf   = lane >> 4;        // half-wave
    const int l16  = lane & 15;

    A += (size_t)blockIdx.z * sA + (size_t)blockIdx.y * 128 * lda;
    B += (size_t)blockIdx.z * sB + (size_t)blockIdx.x * 128 * ldb;

    __shared__ u16 sa[2][128 * 40];    // 32-k rows padded to 40 halfwords
    __shared__ u16 sb[2][128 * 40];

    f32x8 acc[8] = {};

    const int rowL = tid >> 2;         // 0..63
    const int cL   = (tid & 3) * 8;    // 0,8,16,24
    const int soL  = rowL * 40 + cL;
    const int soH  = (rowL + 64) * 40 + cL;

    const u16* gA0 = A + (size_t)rowL * lda + cL;
    const u16* gA1 = A + (size_t)(rowL + 64) * lda + cL;
    const u16* gB0 = B + (size_t)rowL * ldb + cL;
    const u16* gB1 = B + (size_t)(rowL + 64) * ldb + cL;

    const int ktiles = K >> 5;

#if HAVE_ASYNC_LDS
    // ---- async global->LDS double-buffered pipeline (ASYNCcnt tracked) ----
    {
        __builtin_amdgcn_global_load_async_to_lds_b128(GAS1(gA0), LAS3(&sa[0][soL]), 0, 0);
        __builtin_amdgcn_global_load_async_to_lds_b128(GAS1(gA1), LAS3(&sa[0][soH]), 0, 0);
        __builtin_amdgcn_global_load_async_to_lds_b128(GAS1(gB0), LAS3(&sb[0][soL]), 0, 0);
        __builtin_amdgcn_global_load_async_to_lds_b128(GAS1(gB1), LAS3(&sb[0][soH]), 0, 0);
    }
    for (int kt = 0; kt < ktiles; ++kt) {
        const int cur = kt & 1;
        __builtin_amdgcn_s_wait_asynccnt(0);   // my tile-kt LDS writes landed
        __syncthreads();                       // everyone's landed; prev compute done
        if (kt + 1 < ktiles) {
            const int nxt = (kt + 1) & 1;
            const int ko = (kt + 1) * 32;
            __builtin_amdgcn_global_load_async_to_lds_b128(GAS1(gA0 + ko), LAS3(&sa[nxt][soL]), 0, 0);
            __builtin_amdgcn_global_load_async_to_lds_b128(GAS1(gA1 + ko), LAS3(&sa[nxt][soH]), 0, 0);
            __builtin_amdgcn_global_load_async_to_lds_b128(GAS1(gB0 + ko), LAS3(&sb[nxt][soL]), 0, 0);
            __builtin_amdgcn_global_load_async_to_lds_b128(GAS1(gB1 + ko), LAS3(&sb[nxt][soH]), 0, 0);
        }
        FragAB fa[2], fb[4];
#pragma unroll
        for (int mt = 0; mt < 2; ++mt) {
            const int base = (wm * 32 + mt * 16 + l16) * 40 + hf * 8;
            fa[mt].q[0] = *(const uint4*)&sa[cur][base];
            fa[mt].q[1] = *(const uint4*)&sa[cur][base + 16];
        }
#pragma unroll
        for (int nt = 0; nt < 4; ++nt) {
            const int base = (wn * 64 + nt * 16 + l16) * 40 + hf * 8;
            fb[nt].q[0] = *(const uint4*)&sb[cur][base];
            fb[nt].q[1] = *(const uint4*)&sb[cur][base + 16];
        }
#pragma unroll
        for (int mt = 0; mt < 2; ++mt)
#pragma unroll
            for (int nt = 0; nt < 4; ++nt)
                acc[mt * 4 + nt] = __builtin_amdgcn_wmma_f32_16x16x32_bf16(
                    false, fa[mt].v, false, fb[nt].v, (short)0, acc[mt * 4 + nt],
                    false, false);
    }
#else
    // ---- fallback: register-staged double-buffered pipeline ----
    uint4 ra0 = *(const uint4*)gA0;
    uint4 ra1 = *(const uint4*)gA1;
    uint4 rb0 = *(const uint4*)gB0;
    uint4 rb1 = *(const uint4*)gB1;
    for (int kt = 0; kt < ktiles; ++kt) {
        const int cur = kt & 1;
        *(uint4*)&sa[cur][soL] = ra0;
        *(uint4*)&sa[cur][soH] = ra1;
        *(uint4*)&sb[cur][soL] = rb0;
        *(uint4*)&sb[cur][soH] = rb1;
        __syncthreads();
        if (kt + 1 < ktiles) {
            const int ko = (kt + 1) * 32;
            ra0 = *(const uint4*)(gA0 + ko);
            ra1 = *(const uint4*)(gA1 + ko);
            rb0 = *(const uint4*)(gB0 + ko);
            rb1 = *(const uint4*)(gB1 + ko);
            if (kt + 2 < ktiles) {
                __builtin_prefetch(gA0 + ko + 32, 0, 0);
                __builtin_prefetch(gB0 + ko + 32, 0, 0);
            }
        }
        FragAB fa[2], fb[4];
#pragma unroll
        for (int mt = 0; mt < 2; ++mt) {
            const int base = (wm * 32 + mt * 16 + l16) * 40 + hf * 8;
            fa[mt].q[0] = *(const uint4*)&sa[cur][base];
            fa[mt].q[1] = *(const uint4*)&sa[cur][base + 16];
        }
#pragma unroll
        for (int nt = 0; nt < 4; ++nt) {
            const int base = (wn * 64 + nt * 16 + l16) * 40 + hf * 8;
            fb[nt].q[0] = *(const uint4*)&sb[cur][base];
            fb[nt].q[1] = *(const uint4*)&sb[cur][base + 16];
        }
#pragma unroll
        for (int mt = 0; mt < 2; ++mt)
#pragma unroll
            for (int nt = 0; nt < 4; ++nt)
                acc[mt * 4 + nt] = __builtin_amdgcn_wmma_f32_16x16x32_bf16(
                    false, fa[mt].v, false, fb[nt].v, (short)0, acc[mt * 4 + nt],
                    false, false);
    }
#endif

    // ------------------------- epilogue -------------------------
#pragma unroll
    for (int mt = 0; mt < 2; ++mt) {
#pragma unroll
        for (int nt = 0; nt < 4; ++nt) {
            const f32x8 a = acc[mt * 4 + nt];
#pragma unroll
            for (int j = 0; j < 8; ++j) {
                const int m = blockIdx.y * 128 + wm * 32 + mt * 16 + j + hf * 8;
                const int n = blockIdx.x * 128 + wn * 64 + nt * 16 + l16;
                float v = a[j];
                if (epi == 0) {
                    float* Cf = (float*)C + (size_t)blockIdx.z * sC;
                    Cf[(size_t)m * ldc + n] = v;
                } else if (epi == 1) {
                    u16* Cs = (u16*)C + (size_t)blockIdx.z * sC;
                    Cs[(size_t)m * ldc + n] = f2bf(v);
                } else if (epi == 2) {
                    float* Cf = (float*)C + (size_t)blockIdx.z * sC;
                    float o = v * scale + bias[(size_t)m * SEQ + n];
                    if (n > m) o += -1.0e10f;
                    Cf[(size_t)m * ldc + n] = o;
                } else if (epi == 3) {
                    u16* Cs = (u16*)C + (size_t)blockIdx.z * sC;
                    Cs[(size_t)m * ldc + n] = f2bf(geluf(v + bias[n]));
                } else { // epi == 4: accumulate residual path
                    float* Cf = (float*)C + (size_t)blockIdx.z * sC;
                    Cf[(size_t)m * ldc + n] += v + bias[n];
                }
            }
        }
    }
}

// ---------------------------------------------------------------------------
// Host-side orchestration
// ---------------------------------------------------------------------------
extern "C" void kernel_launch(void* const* d_in, const int* in_sizes, int n_in,
                              void* d_out, int out_size, void* d_ws, size_t ws_size,
                              hipStream_t stream) {
    (void)in_sizes; (void)n_in; (void)out_size; (void)ws_size;
    const float* x    = (const float*)d_in[0];
    const float* bias = (const float*)d_in[1];
    const float* lsc  = (const float*)d_in[2];
    const float* loff = (const float*)d_in[3];
    const float* wq   = (const float*)d_in[4];
    const float* wk   = (const float*)d_in[5];
    const float* wv   = (const float*)d_in[6];
    const float* wo   = (const float*)d_in[7];
    const float* w1   = (const float*)d_in[8];
    const float* b1   = (const float*)d_in[9];
    const float* w2   = (const float*)d_in[10];
    const float* b2   = (const float*)d_in[11];
    float* out = (float*)d_out;

    char* ws = (char*)d_ws;
    size_t o = 0;
    auto take = [&](size_t bytes) { size_t r = o; o += (bytes + 255) & ~(size_t)255; return r; };
    const size_t SZ_DM2_BF  = (size_t)DMODEL * DMODEL * 2;   // 33.5 MB
    const size_t SZ_ACT_BF  = (size_t)SEQ * DMODEL * 2;      // 16.8 MB

    const size_t oWQT = take(SZ_DM2_BF);
    const size_t oWKT = take(SZ_DM2_BF);
    const size_t oWVT = take(SZ_DM2_BF);
    const size_t oWOT = take(SZ_DM2_BF);
    const size_t oW1T = take((size_t)DMODEL * DFF * 2);
    const size_t oW2T = take((size_t)DMODEL * DFF * 2);
    const size_t oXN  = take(SZ_ACT_BF);
    const size_t oQR  = take(SZ_ACT_BF);
    const size_t oKR  = take(SZ_ACT_BF);
    const size_t oVR  = take(SZ_ACT_BF);
    const size_t oQP  = take(SZ_ACT_BF);
    const size_t oKP  = take(SZ_ACT_BF);
    const size_t oVT  = take(SZ_ACT_BF);
    const size_t oLOG = take((size_t)NHEADS * SEQ * SEQ * 4);
    const size_t oPRB = take((size_t)NHEADS * SEQ * SEQ * 2);
    const size_t oAV  = take(SZ_ACT_BF);
    const size_t oH   = take((size_t)SEQ * DFF * 2);

    u16* XN  = (u16*)(ws + oXN);
    u16* WQT = (u16*)(ws + oWQT);
    u16* WKT = (u16*)(ws + oWKT);
    u16* WVT = (u16*)(ws + oWVT);
    u16* WOT = (u16*)(ws + oWOT);
    u16* W1T = (u16*)(ws + oW1T);
    u16* W2T = (u16*)(ws + oW2T);
    u16* QR  = (u16*)(ws + oQR);
    u16* KR  = (u16*)(ws + oKR);
    u16* VR  = (u16*)(ws + oVR);
    u16* QP  = (u16*)(ws + oQP);
    u16* KP  = (u16*)(ws + oKP);
    u16* VT  = (u16*)(ws + oVT);
    float* LOG = (float*)(ws + oLOG);
    u16* PRB = (u16*)(ws + oPRB);
    u16* AV  = (u16*)(ws + oAV);
    u16* H   = (u16*)(ws + oH);

    // 1) LayerNorm -> bf16 activations
    ln_kernel<<<SEQ, 256, 0, stream>>>(x, lsc, loff, XN);

    // 2) Weight convert + transpose (f32 [K][N] -> bf16 [N][K])
    wtr_f32_bf16<<<dim3(DMODEL / 32, DMODEL / 32), 256, 0, stream>>>(wq, WQT, DMODEL, DMODEL);
    wtr_f32_bf16<<<dim3(DMODEL / 32, DMODEL / 32), 256, 0, stream>>>(wk, WKT, DMODEL, DMODEL);
    wtr_f32_bf16<<<dim3(DMODEL / 32, DMODEL / 32), 256, 0, stream>>>(wv, WVT, DMODEL, DMODEL);
    wtr_f32_bf16<<<dim3(DMODEL / 32, DMODEL / 32), 256, 0, stream>>>(wo, WOT, DMODEL, DMODEL);
    wtr_f32_bf16<<<dim3(DFF / 32, DMODEL / 32), 256, 0, stream>>>(w1, W1T, DMODEL, DFF);
    wtr_f32_bf16<<<dim3(DMODEL / 32, DFF / 32), 256, 0, stream>>>(w2, W2T, DFF, DMODEL);

    // 3) Q/K/V projections (bf16 out)
    gemm_bf16<<<dim3(DMODEL / 128, SEQ / 128, 1), 256, 0, stream>>>(
        XN, DMODEL, 0, WQT, DMODEL, 0, QR, DMODEL, 0, DMODEL, 1, nullptr, 1.0f);
    gemm_bf16<<<dim3(DMODEL / 128, SEQ / 128, 1), 256, 0, stream>>>(
        XN, DMODEL, 0, WKT, DMODEL, 0, KR, DMODEL, 0, DMODEL, 1, nullptr, 1.0f);
    gemm_bf16<<<dim3(DMODEL / 128, SEQ / 128, 1), 256, 0, stream>>>(
        XN, DMODEL, 0, WVT, DMODEL, 0, VR, DMODEL, 0, DMODEL, 1, nullptr, 1.0f);

    // 4) RoPE on Q,K; transpose V per head -> [h][d][t]
    rope_kernel<<<(SEQ * DMODEL) / 256, 256, 0, stream>>>(QR, QP);
    rope_kernel<<<(SEQ * DMODEL) / 256, 256, 0, stream>>>(KR, KP);
    tr_u16<<<dim3(DHEAD / 32, SEQ / 32, NHEADS), 256, 0, stream>>>(
        VR, DMODEL, DHEAD, VT, SEQ, (long long)DHEAD * SEQ);

    // 5) logits[h] = (Q K^T)/16 + bias + causal  (batched over heads)
    gemm_bf16<<<dim3(SEQ / 128, SEQ / 128, NHEADS), 256, 0, stream>>>(
        QP, DMODEL, DHEAD, KP, DMODEL, DHEAD,
        LOG, SEQ, (long long)SEQ * SEQ, DHEAD, 2, bias, 0.0625f);

    // 6) softmax rows -> bf16 probs
    softmax_kernel<<<NHEADS * SEQ, 256, 0, stream>>>(LOG, PRB);

    // 7) attn_vec[h] = P V   (V^T stored [d][t])
    gemm_bf16<<<dim3(DHEAD / 128, SEQ / 128, NHEADS), 256, 0, stream>>>(
        PRB, SEQ, (long long)SEQ * SEQ, VT, SEQ, (long long)DHEAD * SEQ,
        AV, DMODEL, DHEAD, SEQ, 1, nullptr, 1.0f);

    // 8) attn_out = attn_vec @ wo   (f32 -> d_out)
    gemm_bf16<<<dim3(DMODEL / 128, SEQ / 128, 1), 256, 0, stream>>>(
        AV, DMODEL, 0, WOT, DMODEL, 0, out, DMODEL, 0, DMODEL, 0, nullptr, 1.0f);

    // 9) H = gelu(xn @ w1 + b1)  (bf16)
    gemm_bf16<<<dim3(DFF / 128, SEQ / 128, 1), 256, 0, stream>>>(
        XN, DMODEL, 0, W1T, DMODEL, 0, H, DFF, 0, DMODEL, 3, b1, 1.0f);

    // 10) d_out += H @ w2 + b2
    gemm_bf16<<<dim3(DMODEL / 128, SEQ / 128, 1), 256, 0, stream>>>(
        H, DFF, 0, W2T, DFF, 0, out, DMODEL, 0, DFF, 4, b2, 1.0f);
}
